// Segment_11020886081803
// MI455X (gfx1250) — compile-verified, matrix operand order checked by reading, not run on hardware
//
#include <hip/hip_runtime.h>
#include <stdint.h>

// CDNA5 wave32 WMMA types (probe-confirmed signatures)
typedef __attribute__((ext_vector_type(16))) __bf16 v16bf;
typedef __attribute__((ext_vector_type(8)))  float  v8f;

#define N_SAMP  2048
#define IN_F    32
#define SEG_N   16
#define OUT_F   64
#define KHALF   (IN_F * SEG_N)   // 512  (slope part of K)
#define KDIM    (2 * KHALF)      // 1024 (slope + intercept)
#define ROWS_WG 128

// Map a (K,o) element of W into the per-lane contiguous B-fragment layout for
// V_WMMA_F32_16X16X32_BF16 (CDNA5 ISA, 16-bit operand layout):
//   chunk c = K/32; within chunk, lane-group g = bit3 of K%32 selects
//   K in {0..7,16..23} (g=0) vs {8..15,24..31} (g=1); element j in [0,16).
// Fragment for (c, col=o, g) is 16 consecutive bf16 (32 B, 32-B aligned).
__device__ __forceinline__ int swz_off(int K, int o) {
  int c  = K >> 5;
  int kk = K & 31;
  int g  = (kk >> 3) & 1;
  int j  = (kk & 7) | ((kk >> 4) << 3);
  return ((c * OUT_F + o) * 2 + g) * 16 + j;
}

// Precompute slope/intercept table W[1024][64] in bf16, pre-swizzled into
// d_ws so the GEMM's B-fragment load is one contiguous 32-byte read per lane.
__global__ void prep_w_kernel(const float* __restrict__ x_param,
                              const float* __restrict__ y_param,
                              __bf16* __restrict__ wswz) {
  int p = blockIdx.x * blockDim.x + threadIdx.x;   // one (k,o) pair per thread
  if (p >= KHALF * OUT_F) return;
  int o = p & (OUT_F - 1);
  int k = p >> 6;                // k = i*16 + s
  int i = k >> 4;
  int s = k & 15;
  int base = (i * (SEG_N + 1) + s) * OUT_F + o;
  float xl = x_param[base];
  float xh = x_param[base + OUT_F];
  float y0 = y_param[base];
  float y1 = y_param[base + OUT_F];
  float dv = xh - xl;
  dv = (dv == 0.0f) ? 1.0e-4f : dv;
  float slope = (y1 - y0) / dv;
  float icept = y0 - slope * xl;       // ratio*(x - xlo) + ylo = slope*x + icept
  wswz[swz_off(k, o)]         = (__bf16)slope;
  wswz[swz_off(k + KHALF, o)] = (__bf16)icept;
}

// out[2048 x 64] = A[2048 x 1024] * W[1024 x 64], A synthesized in registers
// from the one-hot segment structure. Intercept chunk c+16 shares its one-hot
// mask with slope chunk c (values 1.0 instead of x), so one mask feeds 2 A
// fragments and 8 WMMAs per iteration. 8 wave32/WG; each wave owns 16x64 of D.
__global__ __launch_bounds__(256) void spline_gemm_kernel(
    const float* __restrict__ x_in,
    const float* __restrict__ x_param,
    const __bf16* __restrict__ wswz,
    float* __restrict__ out) {
  __shared__ float                       grid_s[IN_F][SEG_N + 1];  // 2176 B
  __shared__ __align__(16) float         x_s[ROWS_WG][IN_F];       // 16 KB
  __shared__ __align__(4)  unsigned char seg_s[ROWS_WG][IN_F];     // 4 KB

  const int t    = threadIdx.x;
  const int row0 = blockIdx.x * ROWS_WG;

  // Stage grid breakpoints (column o=0; x_param is o-broadcast in this model).
  for (int p = t; p < IN_F * (SEG_N + 1); p += 256) {
    int i = p / (SEG_N + 1);
    int s = p - i * (SEG_N + 1);
    grid_s[i][s] = x_param[(i * (SEG_N + 1) + s) * OUT_F];
  }

  // Stage this WG's 128x32 x-tile via CDNA5 async copy (ASYNCcnt-tracked):
  // per-lane GLOBAL_LOAD_ASYNC_TO_LDS_B128, 512 B per wave32 instruction.
  {
    unsigned lds_base = (unsigned)(uintptr_t)(&x_s[0][0]);
    unsigned long long gbase =
        (unsigned long long)(uintptr_t)(x_in + (size_t)row0 * IN_F);
#pragma unroll
    for (int it = 0; it < (ROWS_WG * IN_F) / (256 * 4); ++it) {
      int p = (it * 256 + t) * 4;                       // float index
      unsigned           lds_addr = lds_base + (unsigned)p * 4u;
      unsigned long long gaddr    = gbase + (unsigned long long)p * 4ull;
      asm volatile("global_load_async_to_lds_b128 %0, %1, off"
                   :: "v"(lds_addr), "v"(gaddr) : "memory");
    }
    asm volatile("s_wait_asynccnt 0x0" ::: "memory");
  }
  __syncthreads();

  // Exact (f32) segment selection: s* = #{ b in [1,15] : x >= grid[b] }.
  // First/last segments naturally absorb the out-of-range extensions.
  for (int p = t; p < ROWS_WG * IN_F; p += 256) {
    int r = p >> 5;
    int i = p & 31;
    float xe  = x_s[r][i];
    int   cnt = 0;
#pragma unroll
    for (int b = 1; b <= SEG_N - 1; ++b) cnt += (xe >= grid_s[i][b]) ? 1 : 0;
    seg_s[r][i] = (unsigned char)cnt;
  }
  __syncthreads();

  const int lane = t & 31;
  const int wave = t >> 5;
  const int g    = lane >> 4;      // lane group (K-subset for A/B, row-offset for C/D)
  const int n    = lane & 15;      // A: row-in-tile; B/C/D: column-in-tile
  const int rr   = wave * 16 + n;  // A-row (within WG) owned by this lane

  const float*         xrow = x_s[rr];
  const unsigned char* srow = seg_s[rr];

  v8f acc0 = {}, acc1 = {}, acc2 = {}, acc3 = {};

  const __bf16 one  = (__bf16)1.0f;
  const __bf16 zero = (__bf16)0.0f;

  for (int c = 0; c < 16; ++c) {
    // Slope-chunk c covers features i = {2c, 2c+1}, segment window [8g, 8g+8)
    // in each K-half; intercept-chunk c+16 has the identical one-hot mask.
    float2 xv = *(const float2*)&xrow[2 * c];
    unsigned short sp = *(const unsigned short*)&srow[2 * c];
    int s0 = (int)(sp & 0xff) - 8 * g;        // relative to window start
    int s1 = (int)(sp >> 8)   - 8 * g;
    __bf16 x0 = (__bf16)xv.x;
    __bf16 x1 = (__bf16)xv.y;

    // Prefetch next iteration's W region (global_prefetch_b8, L2-resident).
    __builtin_prefetch(wswz + (((c + 1) & 15) * OUT_F * 2) * 16, 0, 1);

    // B fragments: one contiguous 32-byte load per (chunk, column tile).
    const __bf16* wbS = wswz + ((c * OUT_F + n) * 2 + g) * 16;          // slope
    const __bf16* wbI = wswz + (((c + 16) * OUT_F + n) * 2 + g) * 16;   // intercept
    v16bf bS0 = *(const v16bf*)(wbS + 0 * 512);
    v16bf bS1 = *(const v16bf*)(wbS + 1 * 512);
    v16bf bS2 = *(const v16bf*)(wbS + 2 * 512);
    v16bf bS3 = *(const v16bf*)(wbS + 3 * 512);
    v16bf bI0 = *(const v16bf*)(wbI + 0 * 512);
    v16bf bI1 = *(const v16bf*)(wbI + 1 * 512);
    v16bf bI2 = *(const v16bf*)(wbI + 2 * 512);
    v16bf bI3 = *(const v16bf*)(wbI + 3 * 512);

    // Build both A fragments branch-free (v_cndmask chains).
    v16bf ax, a1;
#pragma unroll
    for (int j = 0; j < 8; ++j) {
      bool m0 = (s0 == j);
      bool m1 = (s1 == j);
      ax[j]     = m0 ? x0  : zero;
      ax[8 + j] = m1 ? x1  : zero;
      a1[j]     = m0 ? one : zero;
      a1[8 + j] = m1 ? one : zero;
    }

    acc0 = __builtin_amdgcn_wmma_f32_16x16x32_bf16(false, ax, false, bS0, (short)0, acc0, false, false);
    acc1 = __builtin_amdgcn_wmma_f32_16x16x32_bf16(false, ax, false, bS1, (short)0, acc1, false, false);
    acc2 = __builtin_amdgcn_wmma_f32_16x16x32_bf16(false, ax, false, bS2, (short)0, acc2, false, false);
    acc3 = __builtin_amdgcn_wmma_f32_16x16x32_bf16(false, ax, false, bS3, (short)0, acc3, false, false);
    acc0 = __builtin_amdgcn_wmma_f32_16x16x32_bf16(false, a1, false, bI0, (short)0, acc0, false, false);
    acc1 = __builtin_amdgcn_wmma_f32_16x16x32_bf16(false, a1, false, bI1, (short)0, acc1, false, false);
    acc2 = __builtin_amdgcn_wmma_f32_16x16x32_bf16(false, a1, false, bI2, (short)0, acc2, false, false);
    acc3 = __builtin_amdgcn_wmma_f32_16x16x32_bf16(false, a1, false, bI3, (short)0, acc3, false, false);
  }

  // C/D layout: lane holds column n; VGPR r holds row r + 8*g of the tile.
#pragma unroll
  for (int r = 0; r < 8; ++r) {
    int ro = (row0 + wave * 16 + 8 * g + r) * OUT_F + n;
    out[ro +  0] = acc0[r];
    out[ro + 16] = acc1[r];
    out[ro + 32] = acc2[r];
    out[ro + 48] = acc3[r];
  }
}

extern "C" void kernel_launch(void* const* d_in, const int* in_sizes, int n_in,
                              void* d_out, int out_size, void* d_ws, size_t ws_size,
                              hipStream_t stream) {
  (void)in_sizes; (void)n_in; (void)out_size; (void)ws_size;
  const float* x_in    = (const float*)d_in[0];
  const float* x_param = (const float*)d_in[1];
  const float* y_param = (const float*)d_in[2];
  float*       out     = (float*)d_out;
  __bf16*      wswz    = (__bf16*)d_ws;   // needs KDIM*OUT_F*2 = 128 KB scratch

  prep_w_kernel<<<(KHALF * OUT_F + 255) / 256, 256, 0, stream>>>(x_param, y_param, wswz);
  spline_gemm_kernel<<<N_SAMP / ROWS_WG, 256, 0, stream>>>(x_in, x_param, wswz, out);
}
